// attention_74156905332753
// MI455X (gfx1250) — compile-verified, matrix operand order checked by reading, not run on hardware
//
#include <hip/hip_runtime.h>
#include <hip/hip_bf16.h>

typedef __attribute__((ext_vector_type(16))) __bf16 v16bf;
typedef __attribute__((ext_vector_type(8)))  __bf16 v8bf;
typedef __attribute__((ext_vector_type(8)))  float  v8f;

#define B_SZ   2048
#define L_SZ   64
#define E_SZ   300
#define VPAD   200000      // padding index
#define LP     304         // f32 row stride (LDS / engin)
#define KP     320         // bf16 K padded to 10 WMMA K-steps
#define NPADR  304         // padded N rows in bf16 weight tables
#define NT     19          // N tiles (304/16)

// branch-free tanh: (e^2x - 1)/(e^2x + 1), clamped
__device__ __forceinline__ float fast_tanh(float x) {
    float cx = fminf(fmaxf(x, -10.0f), 10.0f);
    float e  = __expf(2.0f * cx);
    return (e - 1.0f) * __frcp_rn(e + 1.0f);
}

// build a 16x32 WMMA bf16 fragment half per ISA §7.12.2:
// lanes 0-15 take K {kk..kk+7, kk+16..kk+23}, lanes 16-31 take {+8, +24}
__device__ __forceinline__ v16bf ldfrag(const __bf16* row, int kk, int hi) {
    v8bf lo = *(const v8bf*)(row + kk + hi * 8);
    v8bf hh = *(const v8bf*)(row + kk + 16 + hi * 8);
    v16bf r;
    #pragma unroll
    for (int i = 0; i < 8; ++i) { r[i] = lo[i]; r[8 + i] = hh[i]; }
    return r;
}

#define WMMA_BF16(A, Bf, C) \
    __builtin_amdgcn_wmma_f32_16x16x32_bf16(false, (A), false, (Bf), (short)0, (C), false, false)

// ---------------- prep: fp32 [300][300] -> zero-padded bf16 [304][320] ----------------
__global__ void prep_wb_kernel(const float* __restrict__ W, __bf16* __restrict__ Wb) {
    int idx = blockIdx.x * blockDim.x + threadIdx.x;
    if (idx >= NPADR * KP) return;
    int n = idx / KP, k = idx - n * KP;
    float v = (n < E_SZ && k < E_SZ) ? W[n * E_SZ + k] : 0.0f;
    Wb[idx] = (__bf16)v;
}

__global__ void init_loss_kernel(float* loss) { *loss = 0.0f; }

// ---------------- pool: one WG per (pool, batch); writes sen -> d_out + bf16 sen -> ws ----
__global__ void __launch_bounds__(256)
pool_kernel(const int* __restrict__ srcIdx, const int* __restrict__ tgtIdx,
            const float* __restrict__ posT,
            const float* __restrict__ srcEmb, const float* __restrict__ tgtEmb,
            const __bf16* __restrict__ WbMap, const float* __restrict__ Wm1,
            __bf16* __restrict__ senH, float* __restrict__ out) {
    __shared__ float embF[L_SZ * LP];                    // f32 emb (exact weighted sum)
    __shared__ __align__(16) __bf16 embH[L_SZ * KP];     // bf16 emb (WMMA A)
    __shared__ float scoreAcc[L_SZ];
    __shared__ float attn[L_SZ];
    __shared__ float senS[LP];

    const int tid  = threadIdx.x;
    const int pool = blockIdx.x >> 11;
    const int b    = blockIdx.x & (B_SZ - 1);

    const int*   idxP    = pool ? tgtIdx : srcIdx;
    const float* wordEmb = pool ? tgtEmb : srcEmb;

    // ---- gather (float4): emb = word_emb[index] + 0.03*pos, zero-padded ----
    if (tid < L_SZ) scoreAcc[tid] = 0.0f;
    for (int idx = tid; idx < L_SZ * (KP / 4); idx += 256) {
        int m = idx / (KP / 4), e4 = idx - m * (KP / 4);
        float4 v = {0.0f, 0.0f, 0.0f, 0.0f};
        if (e4 < E_SZ / 4) {                             // 75 float4 per row
            int row = idxP[b * L_SZ + m];
            const float4* we = (const float4*)(wordEmb + (size_t)row * E_SZ);
            const float4* pp = (const float4*)(posT + m * E_SZ);
            float4 a = we[e4], p = pp[e4];
            v.x = a.x + 0.03f * p.x;  v.y = a.y + 0.03f * p.y;
            v.z = a.z + 0.03f * p.z;  v.w = a.w + 0.03f * p.w;
        }
        if (e4 < LP / 4) *(float4*)&embF[m * LP + e4 * 4] = v;
        __bf16* eh = &embH[m * KP + e4 * 4];
        eh[0] = (__bf16)v.x; eh[1] = (__bf16)v.y; eh[2] = (__bf16)v.z; eh[3] = (__bf16)v.w;
    }
    __syncthreads();

    // ---- GEMM H = emb @ W_map^T, 4 M-tile accumulators reuse each B fragment 4x ----
    const int wave = tid >> 5, lane = tid & 31, ln16 = lane & 15, hi = lane >> 4;

    for (int nTile = wave; nTile < NT; nTile += 8) {
        // keep A streaming from LDS each iteration (don't let LICM hoist ~320 VGPRs of
        // A fragments out of the loop; B then gets burst-preloaded per iteration instead)
        asm volatile("" ::: "memory");
        if (nTile + 8 < NT)   // warm the next B tile one iteration ahead
            __builtin_prefetch(&WbMap[(size_t)((nTile + 8) * 16 + ln16) * KP], 0, 1);

        v8f c0 = {}, c1 = {}, c2 = {}, c3 = {};
        const __bf16* bRow = &WbMap[(size_t)(nTile * 16 + ln16) * KP];
        const __bf16* a0 = &embH[(0  + ln16) * KP];
        const __bf16* a1 = &embH[(16 + ln16) * KP];
        const __bf16* a2 = &embH[(32 + ln16) * KP];
        const __bf16* a3 = &embH[(48 + ln16) * KP];
        #pragma unroll
        for (int kk = 0; kk < KP; kk += 32) {
            v16bf Bm = ldfrag(bRow, kk, hi);
            c0 = WMMA_BF16(ldfrag(a0, kk, hi), Bm, c0);
            c1 = WMMA_BF16(ldfrag(a1, kk, hi), Bm, c1);
            c2 = WMMA_BF16(ldfrag(a2, kk, hi), Bm, c2);
            c3 = WMMA_BF16(ldfrag(a3, kk, hi), Bm, c3);
        }
        // fused epilogue: scoreAcc[m] += sum_n tanh(h[m][n]) * w1[n]
        int n = nTile * 16 + ln16;
        float w1 = (n < E_SZ) ? Wm1[n] : 0.0f;
        #pragma unroll
        for (int r = 0; r < 8; ++r) {
            float p0 = fast_tanh(c0[r]) * w1;
            float p1 = fast_tanh(c1[r]) * w1;
            float p2 = fast_tanh(c2[r]) * w1;
            float p3 = fast_tanh(c3[r]) * w1;
            p0 += __shfl_xor(p0, 1, 16); p0 += __shfl_xor(p0, 2, 16);
            p0 += __shfl_xor(p0, 4, 16); p0 += __shfl_xor(p0, 8, 16);
            p1 += __shfl_xor(p1, 1, 16); p1 += __shfl_xor(p1, 2, 16);
            p1 += __shfl_xor(p1, 4, 16); p1 += __shfl_xor(p1, 8, 16);
            p2 += __shfl_xor(p2, 1, 16); p2 += __shfl_xor(p2, 2, 16);
            p2 += __shfl_xor(p2, 4, 16); p2 += __shfl_xor(p2, 8, 16);
            p3 += __shfl_xor(p3, 1, 16); p3 += __shfl_xor(p3, 2, 16);
            p3 += __shfl_xor(p3, 4, 16); p3 += __shfl_xor(p3, 8, 16);
            if (ln16 == 0) {
                int mrow = r + hi * 8;
                atomicAdd(&scoreAcc[mrow],      p0);
                atomicAdd(&scoreAcc[mrow + 16], p1);
                atomicAdd(&scoreAcc[mrow + 32], p2);
                atomicAdd(&scoreAcc[mrow + 48], p3);
            }
        }
    }
    __syncthreads();

    // ---- score = tanh(.), pad mask, softmax over seq ----
    if (tid < L_SZ) {
        float s = fast_tanh(scoreAcc[tid]);
        if (idxP[b * L_SZ + tid] == VPAD) s = -INFINITY;
        scoreAcc[tid] = s;
    }
    __syncthreads();
    if (tid == 0) {
        float mx = -INFINITY;
        for (int m = 0; m < L_SZ; ++m) mx = fmaxf(mx, scoreAcc[m]);
        float sm = 0.0f;
        for (int m = 0; m < L_SZ; ++m) { float ev = __expf(scoreAcc[m] - mx); attn[m] = ev; sm += ev; }
        float inv = 1.0f / sm;
        for (int m = 0; m < L_SZ; ++m) attn[m] *= inv;
    }
    __syncthreads();

    // ---- sen = sum_m attn[m] * emb[m] (fp32); write f32 -> d_out, bf16 padded -> ws ----
    for (int e = tid; e < E_SZ; e += 256) {
        float acc = 0.0f;
        for (int m = 0; m < L_SZ; ++m) acc += attn[m] * embF[m * LP + e];
        senS[e] = acc;
    }
    __syncthreads();
    const size_t rowG = (size_t)pool * B_SZ + b;
    for (int k = tid; k < KP; k += 256) {
        float v = (k < E_SZ) ? senS[k] : 0.0f;
        senH[rowG * KP + k] = (__bf16)v;
    }
    for (int e = tid; e < E_SZ; e += 256)
        out[rowG * E_SZ + e] = senS[e];
}

// ---------------- remove: batched GEMM engin = sen @ W_rm^T + b, then projection ----
// 64 WGs: 2 pools x 32 blocks of 64 rows. Reads sen from d_out, overwrites in place.
__global__ void __launch_bounds__(256)
remove_kernel(const __bf16* __restrict__ senH,
              const __bf16* __restrict__ WbSrc, const __bf16* __restrict__ WbTgt,
              const float* __restrict__ bsrc, const float* __restrict__ btgt,
              float* __restrict__ out) {
    __shared__ float enginS[L_SZ * LP];
    __shared__ __align__(16) __bf16 senL[L_SZ * KP];     // staged A rows (40 KB)

    const int tid  = threadIdx.x;
    const int pool = blockIdx.x >> 5;
    const int mBlk = blockIdx.x & 31;
    const __bf16* Wb  = pool ? WbTgt : WbSrc;
    const float*  brm = pool ? btgt  : bsrc;
    const size_t rowBase = (size_t)pool * B_SZ + (size_t)mBlk * L_SZ;

    // stage 64 sen rows into LDS (vectorized 16B copies)
    {
        const uint4* srcp = (const uint4*)(senH + rowBase * KP);
        uint4* dstp = (uint4*)senL;
        for (int i = tid; i < L_SZ * KP / 8; i += 256) dstp[i] = srcp[i];
    }
    __syncthreads();

    const int wave = tid >> 5, lane = tid & 31, ln16 = lane & 15, hi = lane >> 4;

    for (int nTile = wave; nTile < NT; nTile += 8) {
        asm volatile("" ::: "memory");
        if (nTile + 8 < NT)
            __builtin_prefetch(&Wb[(size_t)((nTile + 8) * 16 + ln16) * KP], 0, 1);

        v8f c0 = {}, c1 = {}, c2 = {}, c3 = {};
        const __bf16* bRow = &Wb[(size_t)(nTile * 16 + ln16) * KP];
        const __bf16* a0 = &senL[(0  + ln16) * KP];
        const __bf16* a1 = &senL[(16 + ln16) * KP];
        const __bf16* a2 = &senL[(32 + ln16) * KP];
        const __bf16* a3 = &senL[(48 + ln16) * KP];
        #pragma unroll
        for (int kk = 0; kk < KP; kk += 32) {
            v16bf Bm = ldfrag(bRow, kk, hi);
            c0 = WMMA_BF16(ldfrag(a0, kk, hi), Bm, c0);
            c1 = WMMA_BF16(ldfrag(a1, kk, hi), Bm, c1);
            c2 = WMMA_BF16(ldfrag(a2, kk, hi), Bm, c2);
            c3 = WMMA_BF16(ldfrag(a3, kk, hi), Bm, c3);
        }
        int n = nTile * 16 + ln16;
        float bn = (n < E_SZ) ? brm[n] : 0.0f;
        #pragma unroll
        for (int r = 0; r < 8; ++r) {
            int mrow = r + hi * 8;
            enginS[(mrow)      * LP + n] = c0[r] + bn;
            enginS[(mrow + 16) * LP + n] = c1[r] + bn;
            enginS[(mrow + 32) * LP + n] = c2[r] + bn;
            enginS[(mrow + 48) * LP + n] = c3[r] + bn;
        }
    }
    __syncthreads();

    // per-row: out = sen - engin * (sen.engin)/||engin||^2 (in-place on d_out)
    for (int m = wave; m < L_SZ; m += 8) {
        float* senRow = out + (rowBase + m) * E_SZ;
        float n2 = 0.0f, dt = 0.0f;
        for (int e = lane; e < E_SZ; e += 32) {
            float g = enginS[m * LP + e];
            float s = senRow[e];
            n2 += g * g; dt += g * s;
        }
        #pragma unroll
        for (int off = 16; off >= 1; off >>= 1) {
            n2 += __shfl_xor(n2, off); dt += __shfl_xor(dt, off);
        }
        float scale = dt / n2;
        for (int e = lane; e < E_SZ; e += 32) {
            float g = enginS[m * LP + e];
            senRow[e] = senRow[e] - g * scale;
        }
    }
}

// ---------------- loss: sum_b 100*||src-tgt||^2 / B ----------------
__global__ void loss_kernel(const float* __restrict__ out, float* __restrict__ lossPtr) {
    int b = blockIdx.x;
    const float* s = out;
    const float* t = out + (size_t)B_SZ * E_SZ;
    float acc = 0.0f;
    for (int e = threadIdx.x; e < E_SZ; e += 64) {
        float d = s[(size_t)b * E_SZ + e] - t[(size_t)b * E_SZ + e];
        acc += d * d;
    }
    #pragma unroll
    for (int off = 16; off >= 1; off >>= 1) acc += __shfl_xor(acc, off);
    __shared__ float sh[2];
    if ((threadIdx.x & 31) == 0) sh[threadIdx.x >> 5] = acc;
    __syncthreads();
    if (threadIdx.x == 0)
        atomicAdd(lossPtr, (sh[0] + sh[1]) * (100.0f / (float)B_SZ));
}

extern "C" void kernel_launch(void* const* d_in, const int* in_sizes, int n_in,
                              void* d_out, int out_size, void* d_ws, size_t ws_size,
                              hipStream_t stream) {
    const int*   srcIdx = (const int*)  d_in[0];
    const int*   tgtIdx = (const int*)  d_in[1];
    const float* posT   = (const float*)d_in[2];
    const float* srcEmb = (const float*)d_in[3];
    const float* tgtEmb = (const float*)d_in[4];
    const float* Wmap   = (const float*)d_in[5];
    const float* Wm1    = (const float*)d_in[6];
    const float* Wsrc   = (const float*)d_in[7];
    const float* bsrc   = (const float*)d_in[8];
    const float* Wtgt   = (const float*)d_in[9];
    const float* btgt   = (const float*)d_in[10];
    float* out = (float*)d_out;

    // workspace layout (bf16): 3 weight tables [304][320] + sen [2*2048][320]  (~3.1 MB)
    __bf16* WbMap = (__bf16*)d_ws;
    __bf16* WbSrc = WbMap + (size_t)NPADR * KP;
    __bf16* WbTgt = WbSrc + (size_t)NPADR * KP;
    __bf16* senH  = WbTgt + (size_t)NPADR * KP;

    const int prepBlocks = (NPADR * KP + 255) / 256;
    prep_wb_kernel<<<prepBlocks, 256, 0, stream>>>(Wmap, WbMap);
    prep_wb_kernel<<<prepBlocks, 256, 0, stream>>>(Wsrc, WbSrc);
    prep_wb_kernel<<<prepBlocks, 256, 0, stream>>>(Wtgt, WbTgt);
    init_loss_kernel<<<1, 1, 0, stream>>>(out + (size_t)2 * B_SZ * E_SZ);

    pool_kernel<<<2 * B_SZ, 256, 0, stream>>>(
        srcIdx, tgtIdx, posT, srcEmb, tgtEmb, WbMap, Wm1, senH, out);
    remove_kernel<<<2 * 32, 256, 0, stream>>>(senH, WbSrc, WbTgt, bsrc, btgt, out);
    loss_kernel<<<B_SZ, 64, 0, stream>>>(out, out + (size_t)2 * B_SZ * E_SZ);
}
